// Correction_62534723829924
// MI455X (gfx1250) — matrix operand order repaired and checked by reading, not compiled, not run on hardware
//
#include <hip/hip_runtime.h>
#include <math.h>

// ---------------- problem constants ----------------
#define BB   8
#define CIN  256
#define CH   128      // Cout of conv = C/2
#define HH   128
#define WW   384
#define HWX  (HH*WW)          // 49152
#define SC   (BB*HWX)         // 393216 score pixels
#define LK   1024             // MAX_KPTS
#define KTOT (CIN*9)          // 2304 GEMM K

typedef __attribute__((ext_vector_type(16))) __bf16 v16bf;
typedef __attribute__((ext_vector_type(8)))  float  v8f;

// ---- Tensor Data Mover availability (arity differs across toolchains) ----
#if defined(__has_builtin)
# if __has_builtin(__builtin_amdgcn_tensor_load_to_lds) && __has_builtin(__builtin_amdgcn_s_wait_tensorcnt)
#  define USE_TDM 1
# endif
#endif
#ifndef USE_TDM
# define USE_TDM 0
#endif

#if USE_TDM
typedef __attribute__((ext_vector_type(4))) unsigned int u32x4;
typedef __attribute__((ext_vector_type(8))) int          i32x8;
typedef __attribute__((ext_vector_type(4))) int          i32x4;
#endif

__device__ __forceinline__ unsigned short f2bf(float f) {
    unsigned int u = __float_as_uint(f);
    unsigned int r = u + 0x7FFFu + ((u >> 16) & 1u);
    return (unsigned short)(r >> 16);
}
__device__ __forceinline__ float bf2f(unsigned short h) {
    return __uint_as_float(((unsigned int)h) << 16);
}

// ---------------- K1: f32 -> bf16 convert (feats) ----------------
__global__ void k_f2bf(const float* __restrict__ src, unsigned short* __restrict__ dst, int n4) {
    int g = blockIdx.x * 256 + threadIdx.x;
    if (g >= n4) return;
    float4 f = ((const float4*)src)[g];
    unsigned int lo = (unsigned int)f2bf(f.x) | ((unsigned int)f2bf(f.y) << 16);
    unsigned int hi = (unsigned int)f2bf(f.z) | ((unsigned int)f2bf(f.w) << 16);
    ((uint2*)dst)[g] = make_uint2(lo, hi);
}

// ---------------- K0: pack weights: wpack2[p][cp][n] = bf16pair(w[n][2cp+{0,1}][ky][kx]) ----
__global__ void k_packw(const float* __restrict__ w, unsigned int* __restrict__ wp) {
    int g = blockIdx.x * 256 + threadIdx.x;
    if (g >= 9 * 128 * 128) return;
    int p  = g / (128 * 128);
    int r  = g % (128 * 128);
    int cp = r / 128;
    int n  = r % 128;
    int ky = p / 3, kx = p % 3;
    float w0 = w[(((size_t)n * CIN + 2 * cp    ) * 3 + ky) * 3 + kx];
    float w1 = w[(((size_t)n * CIN + 2 * cp + 1) * 3 + ky) * 3 + kx];
    wp[g] = (unsigned int)f2bf(w0) | ((unsigned int)f2bf(w1) << 16);
}

// ---------------- K2: implicit-GEMM 3x3 conv via WMMA bf16, bias+ReLU fused ------------
__global__ __launch_bounds__(256) void k_conv(const unsigned short* __restrict__ featsbf,
                                              const unsigned int*   __restrict__ wpack2,
                                              const float*          __restrict__ bpa,
                                              unsigned short*       __restrict__ descbf) {
    __shared__ __align__(16) unsigned short As[32 * 72];  // [k][m], padded stride 72
    __shared__ __align__(16) unsigned int   Bs[16 * 128]; // [kpair][n] (2 bf16 K packed per uint)

    const int blk = blockIdx.x;
    const int xt = blk % 6;
    const int y  = (blk / 6) % HH;
    const int b  = blk / (6 * HH);
    const int x0 = xt * 64;

    const int t    = threadIdx.x;
    const int wave = t >> 5;
    const int lane = t & 31;       // wave32
    const int mt   = wave & 3;     // m-tile 0..3 (16 rows each)
    const int nh   = wave >> 2;    // n-half 0..1 (64 couts each)
    const int half = lane >> 4;
    const int m    = lane & 15;

    v8f acc[4] = {};

    const int ca = t >> 3;         // channel 0..31 loaded by this thread
    const int m0 = (t & 7) * 8;    // 8 pixels

    for (int kk = 0; kk < KTOT; kk += 32) {
        const int p  = kk >> 8;          // kernel tap 0..8
        const int c0 = kk & 255;         // channel base
        const int ky = p / 3 - 1, kx = p % 3 - 1;
        const int yy = y + ky;

#if USE_TDM
        // -------- stage B tile via Tensor Data Mover (8KB contiguous) --------
        if (t < 32) {
            unsigned long long gaddr =
                (unsigned long long)(const void*)(wpack2 + ((size_t)p * 128 + (c0 >> 1)) * 128);
            unsigned int lds_off = (unsigned int)(unsigned long long)(const void*)&Bs[0];
            u32x4 g0 = {};
            g0.x = 1u;                                             // count=1, user mode
            g0.y = lds_off;                                        // lds_addr
            g0.z = (unsigned int)(gaddr & 0xFFFFFFFFull);          // global_addr lo
            g0.w = (unsigned int)((gaddr >> 32) & 0x1FFFFFFull)    // global_addr hi (57b)
                 | (2u << 30);                                     // type = 2 ("image")
            i32x8 g1 = {};
            g1[0] = 0x20000;                                       // data_size = 4B
            g1[1] = (int)((2048u & 0xFFFFu) << 16);                // tensor_dim0 lo16 = 2048
            g1[2] = (int)(1u << 16);                               // tensor_dim0 hi=0 | tensor_dim1 lo16 = 1
            g1[3] = (int)(2048u << 16);                            // tile_dim0 = 2048
            g1[4] = 1;                                             // tile_dim1 = 1, tile_dim2 = 0
            g1[5] = 2048;                                          // tensor_dim0_stride lo32
            g1[6] = 0;                                             // stride hi / dim1_stride lo
            g1[7] = 0;
            i32x4 g2 = {};
            i32x4 g3 = {};
#if __clang_major__ >= 23
            i32x8 g4 = {};
            __builtin_amdgcn_tensor_load_to_lds(g0, g1, g2, g3, g4, 0);
#else
            __builtin_amdgcn_tensor_load_to_lds(g0, g1, g2, g3, 0);
#endif
        }
#endif

        // -------- stage A tile (32 k x 64 m) with VALU while TDM runs --------
        unsigned short va[8];
        if ((unsigned)yy < (unsigned)HH) {
            const unsigned short* src =
                featsbf + (((size_t)(b * CIN + c0 + ca)) * HH + yy) * WW;
#pragma unroll
            for (int i = 0; i < 8; ++i) {
                int xx = x0 + m0 + i + kx;
                va[i] = ((unsigned)xx < (unsigned)WW) ? src[xx] : (unsigned short)0;
            }
        } else {
#pragma unroll
            for (int i = 0; i < 8; ++i) va[i] = 0;
        }
        uint4 av;
        av.x = (unsigned int)va[0] | ((unsigned int)va[1] << 16);
        av.y = (unsigned int)va[2] | ((unsigned int)va[3] << 16);
        av.z = (unsigned int)va[4] | ((unsigned int)va[5] << 16);
        av.w = (unsigned int)va[6] | ((unsigned int)va[7] << 16);
        *(uint4*)&As[ca * 72 + m0] = av;

#if !USE_TDM
        // -------- stage B tile (16 kpairs x 128 n), fully coalesced --------
        const unsigned int* bsrc = wpack2 + ((size_t)p * 128 + (c0 >> 1)) * 128;
        uint4 b0v = *(const uint4*)(bsrc + t * 8);
        uint4 b1v = *(const uint4*)(bsrc + t * 8 + 4);
        *(uint4*)&Bs[t * 8]     = b0v;
        *(uint4*)&Bs[t * 8 + 4] = b1v;
#endif

        // prefetch next A rows (gfx1250 global_prefetch_b8)
        if (kk + 32 < KTOT) {
            int kn = kk + 32;
            int pn = kn >> 8, c0n = kn & 255;
            int kyn = pn / 3 - 1;
            int yyn = y + kyn;
            if ((unsigned)yyn < (unsigned)HH)
                __builtin_prefetch(
                    (const void*)(featsbf + (((size_t)(b * CIN + c0n + ca)) * HH + yyn) * WW + x0 + m0),
                    0, 1);
        }

        __syncthreads();           // A staged (and B in fallback path)
#if USE_TDM
        if (t < 32) __builtin_amdgcn_s_wait_tensorcnt(0);   // TDM -> LDS complete
        __syncthreads();           // publish B tile to all waves
#endif

        // -------- A fragment (ISA 16-bit A 16x32 layout) --------
        union { unsigned int u[8]; v16bf v; } afr;
        const int col = mt * 16 + m;
        const int kb  = half * 8;
#pragma unroll
        for (int r = 0; r < 4; ++r) {
            afr.u[r] = (unsigned int)As[(kb + 2 * r) * 72 + col] |
                       ((unsigned int)As[(kb + 2 * r + 1) * 72 + col] << 16);
            afr.u[4 + r] = (unsigned int)As[(kb + 16 + 2 * r) * 72 + col] |
                           ((unsigned int)As[(kb + 16 + 2 * r + 1) * 72 + col] << 16);
        }
        // -------- 4 x (B fragment + WMMA) --------
#pragma unroll
        for (int s = 0; s < 4; ++s) {
            union { unsigned int u[8]; v16bf v; } bfr;
            const int n = nh * 64 + s * 16 + m;
#pragma unroll
            for (int v2 = 0; v2 < 8; ++v2)
                bfr.u[v2] = Bs[((v2 >> 2) * 8 + half * 4 + (v2 & 3)) * 128 + n];
            acc[s] = __builtin_amdgcn_wmma_f32_16x16x32_bf16(
                false, afr.v, false, bfr.v, (short)0, acc[s], false, false);
        }
        __syncthreads();
    }

    // -------- epilogue: bias + ReLU, store bf16 desc (NCHW) --------
#pragma unroll
    for (int s = 0; s < 4; ++s) {
        const int n = nh * 64 + s * 16 + m;
        const float bias = bpa[n];
        unsigned short* dst = descbf + (((size_t)(b * CH + n)) * HH + y) * WW + x0 + mt * 16;
#pragma unroll
        for (int r = 0; r < 8; ++r) {
            int mm = r + 8 * half;
            float v = acc[s][r] + bias;
            v = v > 0.f ? v : 0.f;
            dst[mm] = f2bf(v);
        }
    }
}

// ---------------- K3: 1x1 conv + sigmoid -> scores ----------------
__global__ void k_scores(const unsigned short* __restrict__ descbf,
                         const float* __restrict__ wpb, const float* __restrict__ bpb,
                         float* __restrict__ S) {
    __shared__ float w[CH];
    if (threadIdx.x < CH) w[threadIdx.x] = wpb[threadIdx.x];
    __syncthreads();
    int g = blockIdx.x * 256 + threadIdx.x;
    if (g >= SC) return;
    int b = g / HWX, pix = g % HWX;
    const unsigned short* d = descbf + (size_t)b * CH * HWX + pix;
    float s = 0.f;
    for (int c = 0; c < CH; ++c) s += bf2f(d[(size_t)c * HWX]) * w[c];
    s += bpb[0];
    S[g] = 1.f / (1.f + expf(-s));
}

// ---------------- NMS: separable 9-tap max pools + mask logic ----------------
__global__ void k_hpool(const float* __restrict__ in, float* __restrict__ out) {
    int g = blockIdx.x * 256 + threadIdx.x; if (g >= SC) return;
    int x = g % WW, base = g - x;
    int lo = x - 4 < 0 ? 0 : x - 4;
    int hi = x + 4 >= WW ? WW - 1 : x + 4;
    float mv = -1e30f;
    for (int xx = lo; xx <= hi; ++xx) mv = fmaxf(mv, in[base + xx]);
    out[g] = mv;
}
__global__ void k_vpool(const float* __restrict__ in, float* __restrict__ out) {
    int g = blockIdx.x * 256 + threadIdx.x; if (g >= SC) return;
    int x = g % WW, y = (g / WW) % HH, bh = g / HWX;
    int lo = y - 4 < 0 ? 0 : y - 4;
    int hi = y + 4 >= HH ? HH - 1 : y + 4;
    float mv = -1e30f;
    int base = bh * HWX + x;
    for (int yy = lo; yy <= hi; ++yy) mv = fmaxf(mv, in[base + yy * WW]);
    out[g] = mv;
}
__global__ void k_eq(const float* a, const float* b, float* m) {
    int g = blockIdx.x * 256 + threadIdx.x; if (g >= SC) return;
    m[g] = (a[g] == b[g]) ? 1.f : 0.f;
}
__global__ void k_supp(const float* P, const float* S, float* SMask, float* SS) {
    int g = blockIdx.x * 256 + threadIdx.x; if (g >= SC) return;
    float sm = (P[g] > 0.f) ? 1.f : 0.f;
    SMask[g] = sm;
    SS[g] = (sm != 0.f) ? 0.f : S[g];
}
__global__ void k_upd(float* M, const float* SS, const float* P, const float* SMask) {
    int g = blockIdx.x * 256 + threadIdx.x; if (g >= SC) return;
    bool nm = (SS[g] == P[g]) && (SMask[g] == 0.f);
    M[g] = ((M[g] != 0.f) || nm) ? 1.f : 0.f;
}
__global__ void k_final(const float* M, const float* S, float* NS, float* WK) {
    int g = blockIdx.x * 256 + threadIdx.x; if (g >= SC) return;
    float v = (M[g] != 0.f) ? S[g] : 0.f;
    NS[g] = v; WK[g] = v;
}

// ---------------- top-1024 per batch, deterministic lowest-index tie-break ---------
__global__ void k_topk(float* __restrict__ WK, int* __restrict__ idx) {
    int b = blockIdx.x, t = threadIdx.x;
    __shared__ float sv[256];
    __shared__ int   si[256];
    float* w = WK + (size_t)b * HWX;
    for (int it = 0; it < LK; ++it) {
        float best = -2e38f; int bi = HWX;
        for (int i = t; i < HWX; i += 256) {
            float v = w[i];
            if (v > best || (v == best && i < bi)) { best = v; bi = i; }
        }
        sv[t] = best; si[t] = bi;
        __syncthreads();
        for (int s = 128; s > 0; s >>= 1) {
            if (t < s) {
                float v = sv[t + s]; int j = si[t + s];
                if (v > sv[t] || (v == sv[t] && j < si[t])) { sv[t] = v; si[t] = j; }
            }
            __syncthreads();
        }
        if (t == 0) { idx[b * LK + it] = si[0]; w[si[0]] = -1e38f; }
        __syncthreads();
    }
}

// ---------------- gather keypoint descriptors + L2 normalize ----------------
__global__ void k_kd(const unsigned short* __restrict__ descbf, const int* __restrict__ idx,
                     float* __restrict__ kd) {
    int g = blockIdx.x * 256 + threadIdx.x;
    if (g >= BB * LK) return;
    int b = g >> 10, l = g & (LK - 1);
    int pix = idx[b * LK + l];
    const unsigned short* d = descbf + (size_t)b * CH * HWX + pix;
    float ss = 0.f;
    for (int c = 0; c < CH; ++c) { float v = bf2f(d[(size_t)c * HWX]); ss += v * v; }
    float inv = 1.f / fmaxf(sqrtf(ss), 1e-12f);
    for (int c = 0; c < CH; ++c)
        kd[((size_t)(b * CH + c)) * LK + l] = bf2f(d[(size_t)c * HWX]) * inv;
}

// ---------------- GNN: per-l 8x8 attention across batch ----------------
__global__ __launch_bounds__(64) void k_gnn(const float* __restrict__ kd, float* __restrict__ kd2) {
    int l = blockIdx.x, t = threadIdx.x;
    __shared__ float q[8][CH];
    __shared__ float sc[8][8];
    __shared__ float pr[8][8];
    for (int i = t; i < 8 * CH; i += 64) {
        int n = i >> 7, c = i & (CH - 1);
        q[n][c] = kd[((size_t)(n * CH + c)) * LK + l];
    }
    __syncthreads();
    {
        int n = t >> 3, mm = t & 7;
        float s = 0.f;
        for (int c = 0; c < CH; ++c) s += q[n][c] * q[mm][c];
        sc[n][mm] = s * 0.08838834764831845f;   // 1/sqrt(128)
    }
    __syncthreads();
    if (t < 8) {
        float mx = -1e30f;
        for (int mm = 0; mm < 8; ++mm) mx = fmaxf(mx, sc[t][mm]);
        float e[8], sum = 0.f;
        for (int mm = 0; mm < 8; ++mm) { e[mm] = expf(sc[t][mm] - mx); sum += e[mm]; }
        for (int mm = 0; mm < 8; ++mm) pr[t][mm] = e[mm] / sum;
    }
    __syncthreads();
    {
        int n = t >> 3, c0 = (t & 7) * 16;
        for (int c = c0; c < c0 + 16; ++c) {
            float msg = 0.f;
            for (int mm = 0; mm < 8; ++mm) msg += pr[n][mm] * q[mm][c];
            kd2[((size_t)(n * CH + c)) * LK + l] = 2.f * q[n][c] + msg;   // desc + (desc + msg)
        }
    }
}

// ---------------- projection + min-reduce + theta ----------------
__global__ __launch_bounds__(256) void k_proj(const float* __restrict__ kd2,
                                              const float* __restrict__ wproj,
                                              const float* __restrict__ bproj,
                                              float* __restrict__ theta) {
    int t = threadIdx.x;
    __shared__ float wp[3][CH];
    __shared__ float red[256];
    __shared__ float gm[24];
    for (int i = t; i < 3 * CH; i += 256) wp[i / CH][i % CH] = wproj[i];
    __syncthreads();
    float lmin[8][3];
    for (int b = 0; b < 8; ++b) for (int o = 0; o < 3; ++o) lmin[b][o] = 1e30f;
    for (int l = t; l < LK; l += 256) {
        float p0[3];
        for (int b = 0; b < 8; ++b) {
            float s0 = 0.f, s1 = 0.f, s2 = 0.f;
            for (int c = 0; c < CH; ++c) {
                float v = kd2[((size_t)(b * CH + c)) * LK + l];
                s0 += v * wp[0][c]; s1 += v * wp[1][c]; s2 += v * wp[2][c];
            }
            float pb[3] = { s0 + bproj[0], s1 + bproj[1], s2 + bproj[2] };
            if (b == 0) { p0[0] = pb[0]; p0[1] = pb[1]; p0[2] = pb[2]; }
            for (int o = 0; o < 3; ++o)
                lmin[b][o] = fminf(lmin[b][o], pb[o] - p0[o]);
        }
    }
    for (int bo = 0; bo < 24; ++bo) {
        red[t] = lmin[bo / 3][bo % 3];
        __syncthreads();
        for (int s = 128; s > 0; s >>= 1) {
            if (t < s) red[t] = fminf(red[t], red[t + s]);
            __syncthreads();
        }
        if (t == 0) gm[bo] = red[0];
        __syncthreads();
    }
    if (t == 0) {
        for (int b = 0; b < 8; ++b) {
            float tx = gm[b * 3 + 0], ty = gm[b * 3 + 1], a = gm[b * 3 + 2];  // CORR_RATIO = 1
            float c = cosf(a), s = sinf(a);
            theta[b * 6 + 0] = c;  theta[b * 6 + 1] = -s; theta[b * 6 + 2] = tx;
            theta[b * 6 + 3] = s;  theta[b * 6 + 4] = c;  theta[b * 6 + 5] = ty;
        }
    }
}

// ---------------- affine grid + bilinear sample (zeros padding) ----------------
__global__ __launch_bounds__(128) void k_sample(const float* __restrict__ feats,
                                                const float* __restrict__ theta,
                                                float* __restrict__ out) {
    int blk = blockIdx.x, t = threadIdx.x;
    int xt = blk % 3;
    int y  = (blk / 3) % HH;
    int b  = blk / (3 * HH);
    int x  = xt * 128 + t;

    float gx = (2.f * x + 1.f) / WW - 1.f;
    float gy = (2.f * y + 1.f) / HH - 1.f;
    const float* th = theta + b * 6;
    float u = th[0] * gx + th[1] * gy + th[2];
    float v = th[3] * gx + th[4] * gy + th[5];
    float fx = (u + 1.f) * (WW * 0.5f) - 0.5f;
    float fy = (v + 1.f) * (HH * 0.5f) - 0.5f;
    float x0f = floorf(fx), y0f = floorf(fy);
    float wx1 = fx - x0f, wx0 = 1.f - wx1;
    float wy1 = fy - y0f, wy0 = 1.f - wy1;
    int xi0 = (int)x0f, yi0 = (int)y0f, xi1 = xi0 + 1, yi1 = yi0 + 1;
    float vx0 = (xi0 >= 0 && xi0 < WW) ? 1.f : 0.f;
    float vx1 = (xi1 >= 0 && xi1 < WW) ? 1.f : 0.f;
    float vy0 = (yi0 >= 0 && yi0 < HH) ? 1.f : 0.f;
    float vy1 = (yi1 >= 0 && yi1 < HH) ? 1.f : 0.f;
    int xc0 = xi0 < 0 ? 0 : (xi0 >= WW ? WW - 1 : xi0);
    int xc1 = xi1 < 0 ? 0 : (xi1 >= WW ? WW - 1 : xi1);
    int yc0 = yi0 < 0 ? 0 : (yi0 >= HH ? HH - 1 : yi0);
    int yc1 = yi1 < 0 ? 0 : (yi1 >= HH ? HH - 1 : yi1);
    float w00 = wx0 * wy0 * vx0 * vy0;
    float w10 = wx1 * wy0 * vx1 * vy0;
    float w01 = wx0 * wy1 * vx0 * vy1;
    float w11 = wx1 * wy1 * vx1 * vy1;
    int o00 = yc0 * WW + xc0, o10 = yc0 * WW + xc1;
    int o01 = yc1 * WW + xc0, o11 = yc1 * WW + xc1;
    size_t base = (size_t)b * CIN * HWX;
    int op = y * WW + x;
    for (int c = 0; c < CIN; ++c) {
        const float* img = feats + base + (size_t)c * HWX;
        float r = img[o00] * w00 + img[o10] * w10 + img[o01] * w01 + img[o11] * w11;
        out[base + (size_t)c * HWX + op] = r;
    }
}

// ---------------- launcher ----------------
extern "C" void kernel_launch(void* const* d_in, const int* in_sizes, int n_in,
                              void* d_out, int out_size, void* d_ws, size_t ws_size,
                              hipStream_t stream) {
    const float* feats  = (const float*)d_in[0];
    const float* w_pa   = (const float*)d_in[1];
    const float* b_pa   = (const float*)d_in[2];
    const float* w_pb   = (const float*)d_in[3];
    const float* b_pb   = (const float*)d_in[4];
    const float* w_proj = (const float*)d_in[5];
    const float* b_proj = (const float*)d_in[6];
    float* out = (float*)d_out;

    const size_t FEATS = (size_t)BB * CIN * HWX;   // 100663296
    char* wsb = (char*)d_ws;
    size_t off = 0;
    auto carve = [&](size_t bytes) -> void* {
        void* p = wsb + off;
        off = (off + bytes + 255) & ~(size_t)255;
        return p;
    };
    unsigned short* featsbf = (unsigned short*)carve(FEATS * 2);
    unsigned int*   wpack2  = (unsigned int*)  carve((size_t)9 * 128 * 128 * 4);
    unsigned short* descbf  = (unsigned short*)carve((size_t)BB * CH * HWX * 2);
    float* S     = (float*)carve((size_t)SC * 4);
    float* T     = (float*)carve((size_t)SC * 4);
    float* P     = (float*)carve((size_t)SC * 4);
    float* M     = (float*)carve((size_t)SC * 4);
    float* SMask = (float*)carve((size_t)SC * 4);
    float* SS    = (float*)carve((size_t)SC * 4);
    float* NS    = (float*)carve((size_t)SC * 4);
    float* WK    = (float*)carve((size_t)SC * 4);
    int*   idx   = (int*)  carve((size_t)BB * LK * 4);
    float* kd    = (float*)carve((size_t)BB * CH * LK * 4);
    float* kd2   = (float*)carve((size_t)BB * CH * LK * 4);
    float* theta = (float*)carve(256);

    const int EG = (SC + 255) / 256;

    k_f2bf <<<(int)((FEATS / 4 + 255) / 256), 256, 0, stream>>>(feats, featsbf, (int)(FEATS / 4));
    k_packw<<<(9 * 128 * 128 + 255) / 256, 256, 0, stream>>>(w_pa, wpack2);
    k_conv <<<BB * HH * 6, 256, 0, stream>>>(featsbf, wpack2, b_pa, descbf);
    k_scores<<<EG, 256, 0, stream>>>(descbf, w_pb, b_pb, S);

    // simple_nms
    k_hpool<<<EG, 256, 0, stream>>>(S, T);
    k_vpool<<<EG, 256, 0, stream>>>(T, P);
    k_eq   <<<EG, 256, 0, stream>>>(S, P, M);
    for (int it = 0; it < 2; ++it) {
        k_hpool<<<EG, 256, 0, stream>>>(M, T);
        k_vpool<<<EG, 256, 0, stream>>>(T, P);
        k_supp <<<EG, 256, 0, stream>>>(P, S, SMask, SS);
        k_hpool<<<EG, 256, 0, stream>>>(SS, T);
        k_vpool<<<EG, 256, 0, stream>>>(T, P);
        k_upd  <<<EG, 256, 0, stream>>>(M, SS, P, SMask);
    }
    k_final<<<EG, 256, 0, stream>>>(M, S, NS, WK);

    k_topk<<<BB, 256, 0, stream>>>(WK, idx);
    k_kd  <<<(BB * LK + 255) / 256, 256, 0, stream>>>(descbf, idx, kd);
    k_gnn <<<LK, 64, 0, stream>>>(kd, kd2);
    k_proj<<<1, 256, 0, stream>>>(kd2, w_proj, b_proj, theta);
    k_sample<<<BB * HH * 3, 128, 0, stream>>>(feats, theta, out);
    (void)in_sizes; (void)n_in; (void)out_size; (void)ws_size;
}
